// BiLSTMCRF_60644938219646
// MI455X (gfx1250) — matrix operand order repaired and checked by reading, not compiled, not run on hardware
//
#include <hip/hip_runtime.h>
#include <hip/hip_bf16.h>

#define S_LEN 2048
#define BATCH 64
#define FDIM  768
#define HDIM  128
#define HL    64
#define G4    256
#define NTAG  2

typedef __attribute__((ext_vector_type(16))) __bf16 v16bf;
typedef __attribute__((ext_vector_type(8)))  __bf16 v8bf;
typedef __attribute__((ext_vector_type(8)))  float  v8f;
typedef __attribute__((ext_vector_type(4)))  float  v4f;
typedef __attribute__((ext_vector_type(4)))  unsigned int u32x4;
typedef __attribute__((ext_vector_type(4)))  int    i32x4;
typedef __attribute__((ext_vector_type(8)))  int    i32x8;

__device__ __forceinline__ v8f wmma_bf16(v16bf a, v16bf b, v8f c) {
    return __builtin_amdgcn_wmma_f32_16x16x32_bf16(false, a, false, b, (short)0, c,
                                                   false, false);
}

// Packed B operand: tile = 32 lanes x 16 bf16, contiguous per lane (32B load).
__device__ __forceinline__ v16bf load_bp(const __bf16* p, int tile, int lane) {
    return *(const v16bf*)(p + ((size_t)tile * 32 + lane) * 16);
}

// A operand from row-major bf16 (global or LDS). row_ptr pre-offset to the
// lane's row. 16-bit A layout: lanes 0-15 K=[0..7],[16..23]; lanes 16-31 +8.
__device__ __forceinline__ v16bf load_a_bf16(const __bf16* row_ptr, int k0, int lane) {
    const __bf16* p = row_ptr + k0 + ((lane & 16) ? 8 : 0);
    v8bf c0 = *(const v8bf*)(p);
    v8bf c1 = *(const v8bf*)(p + 16);
    v16bf r;
#pragma unroll
    for (int i = 0; i < 8; ++i) { r[i] = c0[i]; r[8 + i] = c1[i]; }
    return r;
}

// A operand from row-major f32 with on-the-fly bf16 conversion.
__device__ __forceinline__ v16bf load_a_f32(const float* p) {
    v4f q0 = *(const v4f*)(p);
    v4f q1 = *(const v4f*)(p + 4);
    v4f q2 = *(const v4f*)(p + 16);
    v4f q3 = *(const v4f*)(p + 20);
    v16bf a;
#pragma unroll
    for (int i = 0; i < 4; ++i) {
        a[i]      = (__bf16)q0[i];
        a[4 + i]  = (__bf16)q1[i];
        a[8 + i]  = (__bf16)q2[i];
        a[12 + i] = (__bf16)q3[i];
    }
    return a;
}

// TDM: DMA one contiguous 32KB tile (4096 x 8B) from global into LDS.
// D# built per CDNA5 ISA 8.3/8.4: group0 {count=1, lds_addr, global_addr, type=2},
// group1 {data_size=8B, tensor_dim0=4096, tensor_dim1=1, tile_dim0=4096,
//         tile_dim1=1, tensor_dim0_stride=4096}.
__device__ __forceinline__ void tdm_load_xg(const __bf16* gsrc, unsigned int lds_off) {
    unsigned long long ga = (unsigned long long)(size_t)gsrc;
    u32x4 g0;
    g0[0] = 1u;                                        // count=1 (valid D#)
    g0[1] = lds_off;                                   // LDS byte address
    g0[2] = (unsigned int)(ga & 0xffffffffu);          // global_addr[31:0]
    g0[3] = (unsigned int)((ga >> 32) & 0x01ffffffu)   // global_addr[56:32]
          | (2u << 30);                                // type=2 ("image")
    i32x8 g1;
    g1[0] = 0x00030000;          // workgroup_mask=0, data_size=3 (8 bytes)
    g1[1] = (int)(0x1000u << 16);// tensor_dim0[15:0]=4096 at bits[63:48]
    g1[2] = 0x00010000;          // tensor_dim0[31:16]=0, tensor_dim1=1
    g1[3] = (int)(0x1000u << 16);// tile_dim0=4096 at bits[127:112]
    g1[4] = 1;                   // tile_dim1=1, tile_dim2=0
    g1[5] = 0x1000;              // tensor_dim0_stride=4096
    g1[6] = 0;
    g1[7] = 0;
    i32x4 z4 = {0, 0, 0, 0};
#if defined(__clang_major__) && (__clang_major__ >= 23)
    i32x8 z8 = {0, 0, 0, 0, 0, 0, 0, 0};
    __builtin_amdgcn_tensor_load_to_lds(g0, g1, z4, z4, z8, 0);
#else
    __builtin_amdgcn_tensor_load_to_lds(g0, g1, z4, z4, 0);
#endif
}

// ---------------- K0: pack weights into WMMA B-operand order (bf16) ---------
__global__ __launch_bounds__(256)
void k_pack(const float* __restrict__ projW, const float* __restrict__ wihf,
            const float* __restrict__ wihb, const float* __restrict__ whhf,
            const float* __restrict__ whhb, __bf16* __restrict__ packed) {
    int idx = blockIdx.x * 256 + threadIdx.x;
    const int R0 = 24 * 8 * 512;   // proj_W  (768x128)
    const int R1 = 4 * 16 * 512;   // w_ih^T  (128x256) each
    const int R2 = 2 * 16 * 512;   // w_hh^T  (64x256)  each
    if (idx >= R0 + 2 * R1 + 2 * R2) return;
    int e = idx & 511;
    int lane = e >> 4, i = e & 15;
    int kk = (i & 7) + ((lane & 16) ? 8 : 0) + ((i & 8) ? 16 : 0);  // K within 32
    int nn = lane & 15;                   // N within 16
    float v;
    if (idx < R0) {
        int t = idx >> 9;
        int k = (t >> 3) * 32 + kk, n = (t & 7) * 16 + nn;
        v = projW[k * HDIM + n];
    } else if (idx < R0 + R1) {
        int t = (idx - R0) >> 9;
        int k = (t >> 4) * 32 + kk, n = (t & 15) * 16 + nn;
        v = wihf[n * HDIM + k];
    } else if (idx < R0 + 2 * R1) {
        int t = (idx - R0 - R1) >> 9;
        int k = (t >> 4) * 32 + kk, n = (t & 15) * 16 + nn;
        v = wihb[n * HDIM + k];
    } else if (idx < R0 + 2 * R1 + R2) {
        int t = (idx - R0 - 2 * R1) >> 9;
        int k = (t >> 4) * 32 + kk, n = (t & 15) * 16 + nn;
        v = whhf[n * HL + k];
    } else {
        int t = (idx - R0 - 2 * R1 - R2) >> 9;
        int k = (t >> 4) * 32 + kk, n = (t & 15) * 16 + nn;
        v = whhb[n * HL + k];
    }
    packed[idx] = (__bf16)v;
}

// ---------------- K1: projection + LayerNorm -> x bf16 in (S,B,H) ----------
__global__ __launch_bounds__(256)
void k_proj_ln(const float* __restrict__ feat, const __bf16* __restrict__ projP,
               const float* __restrict__ projb, const float* __restrict__ lng,
               const float* __restrict__ lnb, __bf16* __restrict__ xbf) {
    const int lane = threadIdx.x & 31;
    const int wave = threadIdx.x >> 5;
    const int m0 = blockIdx.x * 128 + wave * 16;     // M index = s*64 + b
    const int arow = m0 + (lane & 15);
    const int frow = (arow & 63) * S_LEN + (arow >> 6);   // features (B,S,F)
    const float* abase = feat + (size_t)frow * FDIM + ((lane & 16) ? 8 : 0);
    v8f acc[8] = {};
    // Ping-pong B-tile buffers (no copies, no WMMA->VALU hazard stalls):
    // even kt consumes b0 while b1 prefetches kt+1; odd kt the reverse.
    v16bf b0[8], b1[8];
#pragma unroll
    for (int nt = 0; nt < 8; ++nt) b0[nt] = load_bp(projP, nt, lane);
#pragma unroll 1
    for (int kt = 0; kt < 24; kt += 2) {
#pragma unroll
        for (int nt = 0; nt < 8; ++nt)
            b1[nt] = load_bp(projP, (kt + 1) * 8 + nt, lane);
        v16bf a = load_a_f32(abase + kt * 32);
#pragma unroll
        for (int nt = 0; nt < 8; ++nt)
            acc[nt] = wmma_bf16(a, b0[nt], acc[nt]);
        if (kt + 2 < 24) {
#pragma unroll
            for (int nt = 0; nt < 8; ++nt)
                b0[nt] = load_bp(projP, (kt + 2) * 8 + nt, lane);
        }
        a = load_a_f32(abase + kt * 32 + 32);
#pragma unroll
        for (int nt = 0; nt < 8; ++nt)
            acc[nt] = wmma_bf16(a, b1[nt], acc[nt]);
    }
    float pb[8], gg[8], bb[8];
#pragma unroll
    for (int nt = 0; nt < 8; ++nt) {
        int col = nt * 16 + (lane & 15);
        pb[nt] = projb[col]; gg[nt] = lng[col]; bb[nt] = lnb[col];
    }
    const int rbase = m0 + ((lane & 16) ? 8 : 0);
#pragma unroll
    for (int r = 0; r < 8; ++r) {
        float vals[8], s = 0.f, s2 = 0.f;
#pragma unroll
        for (int nt = 0; nt < 8; ++nt) {
            float v = acc[nt][r] + pb[nt];
            vals[nt] = v; s += v; s2 += v * v;
        }
#pragma unroll
        for (int off = 1; off < 16; off <<= 1) {
            s  += __shfl_xor(s, off);
            s2 += __shfl_xor(s2, off);
        }
        float mean = s * (1.f / 128.f);
        float var  = s2 * (1.f / 128.f) - mean * mean;
        float rstd = rsqrtf(var + 1e-5f);
        __bf16* op = xbf + (size_t)(rbase + r) * HDIM;
#pragma unroll
        for (int nt = 0; nt < 8; ++nt) {
            int col = nt * 16 + (lane & 15);
            op[col] = (__bf16)((vals[nt] - mean) * rstd * gg[nt] + bb[nt]);
        }
    }
}

// ---------------- K2: input gates xg = x @ w_ih^T + b  -> (S,B,256) bf16 ----
__global__ __launch_bounds__(256)
void k_gatex(const __bf16* __restrict__ xbf, const __bf16* __restrict__ wP,
             const float* __restrict__ bias, __bf16* __restrict__ xg, int reversed) {
    const int lane = threadIdx.x & 31;
    const int wave = threadIdx.x >> 5;
    const int m0 = blockIdx.x * 128 + wave * 16;
    const __bf16* arow = xbf + (size_t)(m0 + (lane & 15)) * HDIM;
    v8f acc[16] = {};
#pragma unroll
    for (int kt = 0; kt < 4; ++kt) {
        v16bf a = load_a_bf16(arow, kt * 32, lane);
#pragma unroll
        for (int nt = 0; nt < 16; ++nt)
            acc[nt] = wmma_bf16(a, load_bp(wP, kt * 16 + nt, lane), acc[nt]);
    }
    const int rbase = m0 + ((lane & 16) ? 8 : 0);
#pragma unroll
    for (int nt = 0; nt < 16; ++nt) {
        int col = nt * 16 + (lane & 15);
        float bs = bias[col];
#pragma unroll
        for (int r = 0; r < 8; ++r) {
            int row = rbase + r;
            int orow = reversed ? ((S_LEN - 1 - (row >> 6)) * BATCH + (row & 63)) : row;
            xg[(size_t)orow * G4 + col] = (__bf16)(acc[nt][r] + bs);
        }
    }
}

// ---------------- K3: persistent sequential LSTM (one WG per direction) -----
// TDM double-buffers the next step's 32KB xg tile into LDS while the current
// step's recurrence computes; s_wait_tensorcnt + workgroup barrier fence it.
__global__ __launch_bounds__(256)
void k_lstm(const __bf16* __restrict__ xgf, const __bf16* __restrict__ xgb,
            const __bf16* __restrict__ hhfP, const __bf16* __restrict__ hhbP,
            const float* __restrict__ emW, float* __restrict__ ebuf_f,
            float* __restrict__ ebuf_b) {
    const int dir = blockIdx.x;
    const __bf16* xg  = dir ? xgb : xgf;
    const __bf16* hhP = dir ? hhbP : hhfP;
    float* ebuf = dir ? ebuf_b : ebuf_f;
    const int jbase = dir * HL;

    __shared__ __bf16 xstage[2][BATCH * G4];  // 2 x 32 KB TDM staging
    __shared__ __bf16 gbuf[BATCH * G4];       // 32 KB gates
    __shared__ __bf16 hbuf[BATCH * HL];       //  8 KB hidden state (A operand)
    __shared__ float  epart[4 * BATCH * NTAG];

    const int tid = threadIdx.x;
    const int lane = tid & 31, wave = tid >> 5;
    const int eb = tid & 63, jg = tid >> 6;   // elementwise role: (b, j-group)

    const unsigned int lds0 = (unsigned int)(size_t)(__bf16*)&xstage[0][0];
    const unsigned int lds1 = (unsigned int)(size_t)(__bf16*)&xstage[1][0];

    float cst[16], emw0[16], emw1[16];
#pragma unroll
    for (int jj = 0; jj < 16; ++jj) {
        cst[jj] = 0.f;
        int j = jbase + jg * 16 + jj;
        emw0[jj] = emW[j * NTAG + 0];
        emw1[jj] = emW[j * NTAG + 1];
    }
    if (wave == 0) {
        tdm_load_xg(xg, lds0);                   // stage t=0
        __builtin_amdgcn_s_wait_tensorcnt(0);
    }
    for (int i = tid; i < BATCH * HL; i += 256) hbuf[i] = (__bf16)0.f;
    __syncthreads();

    for (int t = 0; t < S_LEN; ++t) {
        if (wave == 0 && t + 1 < S_LEN)          // overlap DMA of t+1 with step t
            tdm_load_xg(xg + (size_t)(t + 1) * BATCH * G4,
                        ((t + 1) & 1) ? lds1 : lds0);
        const __bf16* xs = xstage[t & 1];

        // recurrent GEMM: g += h @ w_hh^T ; wave owns nt = {2w, 2w+1}, mt = 0..3
        v8f acc[8] = {};
#pragma unroll
        for (int kt = 0; kt < 2; ++kt) {
            v16bf a[4];
#pragma unroll
            for (int mt = 0; mt < 4; ++mt)
                a[mt] = load_a_bf16(hbuf + (size_t)(mt * 16 + (lane & 15)) * HL,
                                    kt * 32, lane);
            v16bf bt[2];
#pragma unroll
            for (int q = 0; q < 2; ++q)
                bt[q] = load_bp(hhP, kt * 16 + wave * 2 + q, lane);
#pragma unroll
            for (int tI = 0; tI < 8; ++tI)
                acc[tI] = wmma_bf16(a[tI & 3], bt[tI >> 2], acc[tI]);
        }
        // spill gates: g = acc + xg_t (from LDS stage)
#pragma unroll
        for (int tI = 0; tI < 8; ++tI) {
            int colb = (wave * 2 + (tI >> 2)) * 16 + (lane & 15);
            int rowb = (tI & 3) * 16 + ((lane & 16) ? 8 : 0);
#pragma unroll
            for (int r = 0; r < 8; ++r) {
                int row = rowb + r;
                float g = acc[tI][r] + (float)xs[row * G4 + colb];
                gbuf[row * G4 + colb] = (__bf16)g;
            }
        }
        __syncthreads();
        // elementwise cell update + emission partials
        float e0 = 0.f, e1 = 0.f;
#pragma unroll
        for (int jj = 0; jj < 16; ++jj) {
            int j = jg * 16 + jj;
            int gb = eb * G4 + j;
            float gi = (float)gbuf[gb];
            float gf = (float)gbuf[gb + 64];
            float gc = (float)gbuf[gb + 128];
            float go = (float)gbuf[gb + 192];
            float si = 1.f / (1.f + __expf(-gi));
            float sf = 1.f / (1.f + __expf(-gf));
            float so = 1.f / (1.f + __expf(-go));
            float c = sf * cst[jj] + si * tanhf(gc);
            cst[jj] = c;
            float h = so * tanhf(c);
            hbuf[eb * HL + j] = (__bf16)h;
            e0 += h * emw0[jj];
            e1 += h * emw1[jj];
        }
        epart[(jg * BATCH + eb) * NTAG + 0] = e0;
        epart[(jg * BATCH + eb) * NTAG + 1] = e1;
        if (wave == 0 && t + 1 < S_LEN)          // stage(t+1) complete before reuse
            __builtin_amdgcn_s_wait_tensorcnt(0);
        __syncthreads();
        if (tid < 128) {
            int b2 = tid & 63, k = tid >> 6;
            float e = epart[(0 * BATCH + b2) * NTAG + k] +
                      epart[(1 * BATCH + b2) * NTAG + k] +
                      epart[(2 * BATCH + b2) * NTAG + k] +
                      epart[(3 * BATCH + b2) * NTAG + k];
            int s = dir ? (S_LEN - 1 - t) : t;
            ebuf[((size_t)b2 * S_LEN + s) * NTAG + k] = e;
        }
    }
}

// ---------------- K4: combine emissions (deterministic, no atomics) ---------
__global__ __launch_bounds__(256)
void k_emis(const float* __restrict__ ef, const float* __restrict__ ebk,
            const float* __restrict__ emb, float* __restrict__ out) {
    int i = blockIdx.x * 256 + threadIdx.x;
    if (i < BATCH * S_LEN * NTAG) out[i] = emb[i & 1] + ef[i] + ebk[i];
}

// ---------------- K5: CRF negative log-likelihood ----------------------------
__global__ __launch_bounds__(64)
void k_crf(const float* __restrict__ emis, const int* __restrict__ labels,
           const float* __restrict__ cs, const float* __restrict__ ce,
           const float* __restrict__ ct, float* __restrict__ loss_out) {
    __shared__ float red[64];
    int b = threadIdx.x;
    const float* e = emis + (size_t)b * S_LEN * NTAG;
    const int* lab = labels + (size_t)b * S_LEN;
    float t00 = ct[0], t01 = ct[1], t10 = ct[2], t11 = ct[3];
    float a0 = cs[0] + e[0], a1 = cs[1] + e[1];
    int yp = lab[0];
    float num = cs[yp] + (yp ? e[1] : e[0]);
    for (int s = 1; s < S_LEN; ++s) {
        float e0 = e[s * 2], e1 = e[s * 2 + 1];
        int y = lab[s];
        num += ct[yp * 2 + y] + (y ? e1 : e0);
        yp = y;
        float m0 = fmaxf(a0 + t00, a1 + t10);
        float n0 = m0 + __logf(__expf(a0 + t00 - m0) + __expf(a1 + t10 - m0)) + e0;
        float m1 = fmaxf(a0 + t01, a1 + t11);
        float n1 = m1 + __logf(__expf(a0 + t01 - m1) + __expf(a1 + t11 - m1)) + e1;
        a0 = n0; a1 = n1;
    }
    num += ce[yp];
    float f0 = a0 + ce[0], f1 = a1 + ce[1];
    float mz = fmaxf(f0, f1);
    float logZ = mz + __logf(__expf(f0 - mz) + __expf(f1 - mz));
    red[b] = logZ - num;
    __syncthreads();
    if (b == 0) {
        float tsum = 0.f;
        for (int i = 0; i < 64; ++i) tsum += red[i];
        *loss_out = tsum;
    }
}

// ---------------- K6: Viterbi decode ----------------------------------------
__global__ __launch_bounds__(64)
void k_viterbi(const float* __restrict__ emis, const float* __restrict__ cs,
               const float* __restrict__ ce, const float* __restrict__ ct,
               unsigned char* __restrict__ bp, float* __restrict__ pred) {
    int b = threadIdx.x;
    const float* e = emis + (size_t)b * S_LEN * NTAG;
    float t00 = ct[0], t01 = ct[1], t10 = ct[2], t11 = ct[3];
    float s0 = cs[0] + e[0], s1 = cs[1] + e[1];
    unsigned char* bpb = bp + b * 2;
    for (int s = 1; s < S_LEN; ++s) {
        float c00 = s0 + t00, c10 = s1 + t10;  // into state 0
        float c01 = s0 + t01, c11 = s1 + t11;  // into state 1
        int b0 = c10 > c00;                     // argmax, first-max tie-break
        int b1 = c11 > c01;
        float n0 = (b0 ? c10 : c00) + e[s * 2];
        float n1 = (b1 ? c11 : c01) + e[s * 2 + 1];
        bpb[(size_t)s * BATCH * 2 + 0] = (unsigned char)b0;
        bpb[(size_t)s * BATCH * 2 + 1] = (unsigned char)b1;
        s0 = n0; s1 = n1;
    }
    int last = (s1 + ce[1] > s0 + ce[0]) ? 1 : 0;
    float* po = pred + (size_t)b * S_LEN;
    po[S_LEN - 1] = (float)last;
    for (int s = S_LEN - 1; s >= 1; --s) {
        last = bpb[(size_t)s * BATCH * 2 + last];
        po[s - 1] = (float)last;
    }
}

extern "C" void kernel_launch(void* const* d_in, const int* in_sizes, int n_in,
                              void* d_out, int out_size, void* d_ws, size_t ws_size,
                              hipStream_t stream) {
    (void)in_sizes; (void)n_in; (void)out_size; (void)ws_size;
    const float* feat   = (const float*)d_in[0];
    const int*   labels = (const int*)  d_in[1];
    const float* projW  = (const float*)d_in[2];
    const float* projb  = (const float*)d_in[3];
    const float* lng    = (const float*)d_in[4];
    const float* lnb    = (const float*)d_in[5];
    const float* wihf   = (const float*)d_in[6];
    const float* whhf   = (const float*)d_in[7];
    const float* bf     = (const float*)d_in[8];
    const float* wihb   = (const float*)d_in[9];
    const float* whhb   = (const float*)d_in[10];
    const float* bb     = (const float*)d_in[11];
    const float* emW    = (const float*)d_in[12];
    const float* emb    = (const float*)d_in[13];
    const float* cs     = (const float*)d_in[14];
    const float* ce     = (const float*)d_in[15];
    const float* ct     = (const float*)d_in[16];

    char* ws = (char*)d_ws;
    size_t off = 0;
    __bf16* packed = (__bf16*)(ws + off); off += (size_t)196608 * 2;
    __bf16* xbf  = (__bf16*)(ws + off); off += (size_t)S_LEN * BATCH * HDIM * 2;
    __bf16* xgf  = (__bf16*)(ws + off); off += (size_t)S_LEN * BATCH * G4 * 2;
    __bf16* xgb  = (__bf16*)(ws + off); off += (size_t)S_LEN * BATCH * G4 * 2;
    float*  ebf  = (float*)(ws + off);  off += (size_t)BATCH * S_LEN * NTAG * 4;
    float*  ebb  = (float*)(ws + off);  off += (size_t)BATCH * S_LEN * NTAG * 4;
    unsigned char* bp = (unsigned char*)(ws + off); off += (size_t)S_LEN * BATCH * 2;

    const __bf16* projP = packed;
    const __bf16* ihfP  = packed + 98304;
    const __bf16* ihbP  = packed + 131072;
    const __bf16* hhfP  = packed + 163840;
    const __bf16* hhbP  = packed + 180224;

    float* emis_out = (float*)d_out;
    float* loss_out = emis_out + BATCH * S_LEN * NTAG;
    float* pred_out = loss_out + 1;

    k_pack<<<768, 256, 0, stream>>>(projW, wihf, wihb, whhf, whhb, packed);
    k_proj_ln<<<(S_LEN * BATCH) / 128, 256, 0, stream>>>(feat, projP, projb, lng, lnb, xbf);
    k_gatex<<<(S_LEN * BATCH) / 128, 256, 0, stream>>>(xbf, ihfP, bf, xgf, 0);
    k_gatex<<<(S_LEN * BATCH) / 128, 256, 0, stream>>>(xbf, ihbP, bb, xgb, 1);
    k_lstm<<<2, 256, 0, stream>>>(xgf, xgb, hhfP, hhbP, emW, ebf, ebb);
    k_emis<<<(BATCH * S_LEN * NTAG + 255) / 256, 256, 0, stream>>>(ebf, ebb, emb, emis_out);
    k_crf<<<1, 64, 0, stream>>>(emis_out, labels, cs, ce, ct, loss_out);
    k_viterbi<<<1, 64, 0, stream>>>(emis_out, cs, ce, ct, bp, pred_out);
}
